// PreActBlock_conv_Q_69303592288763
// MI455X (gfx1250) — compile-verified, matrix operand order checked by reading, not run on hardware
//
#include <hip/hip_runtime.h>

// ---------------------------------------------------------------------------
// PreAct residual block with 1-bit (DoReFa) weights, CDNA5 / gfx1250.
// Convs run as implicit GEMM on v_wmma_f32_16x16x32_bf16.
// 128x128 block tile, double-buffered LDS, 8 WMMAs per wave per K-step,
// async global->LDS staging (ASYNCcnt) when the toolchain provides it.
// ---------------------------------------------------------------------------

typedef __attribute__((ext_vector_type(16))) __bf16 v16bf;
typedef __attribute__((ext_vector_type(8)))  float  v8f;
typedef __attribute__((ext_vector_type(4)))  int    v4i;
typedef unsigned short u16;

#define BATCH 32
#define CH    256
#define IMGH  56
#define IMGW  56
#define HW    3136            // 56*56
#define NPIX  (BATCH * HW)    // 100352
#define KDIM  (9 * CH)        // 2304 ; k = tap*256 + c
#define BN_EPS 1e-5f

// conv tiling
#define BM 128                // output channels per block
#define BN_TILE 128           // pixels per block
#define BK 32                 // K per step (one WMMA K)
#define LDS_K 40              // row stride in halfwords: 80B (16B aligned, conflict-free)

// ---- async global->LDS staging support ------------------------------------
#if defined(__has_builtin)
#if __has_builtin(__builtin_amdgcn_global_load_async_to_lds_b128)
#define HAVE_ASYNC_LDS 1
#endif
#endif

__device__ __forceinline__ void cp_async16(const u16* __restrict__ g,
                                           u16* __restrict__ l) {
#ifdef HAVE_ASYNC_LDS
  // emits global_load_async_to_lds_b128 (tracked by ASYNCcnt)
  __builtin_amdgcn_global_load_async_to_lds_b128((v4i*)g, (v4i*)l, 0, 0);
#else
  *(uint4*)l = *(const uint4*)g;
#endif
}

__device__ __forceinline__ void async_fence() {
#ifdef HAVE_ASYNC_LDS
#if __has_builtin(__builtin_amdgcn_s_wait_asynccnt)
  __builtin_amdgcn_s_wait_asynccnt(0);
#else
  asm volatile("s_wait_asynccnt 0x0" ::: "memory");
#endif
#endif
}

__device__ __forceinline__ u16 f2bf(float f) {
  unsigned u = __float_as_uint(f);
  unsigned r = (u + 0x7FFFu + ((u >> 16) & 1u)) >> 16;
  return (u16)r;
}
__device__ __forceinline__ float bf2f(u16 h) {
  return __uint_as_float(((unsigned)h) << 16);
}

// --------------------------- BN statistics ---------------------------------

// One block per (n,c) plane of NCHW fp32 input; fixed-order tree reduction.
__global__ __launch_bounds__(256) void bn0_partial_kernel(
    const float* __restrict__ x, float* __restrict__ ps, float* __restrict__ pq) {
  __shared__ float ss[256], sq[256];
  const int b = blockIdx.x;                 // n*CH + c
  const float* p = x + (size_t)b * HW;
  float s = 0.f, q = 0.f;
  for (int i = threadIdx.x; i < HW; i += 256) { float v = p[i]; s += v; q += v * v; }
  ss[threadIdx.x] = s; sq[threadIdx.x] = q;
  __syncthreads();
  for (int st = 128; st > 0; st >>= 1) {
    if ((int)threadIdx.x < st) {
      ss[threadIdx.x] += ss[threadIdx.x + st];
      sq[threadIdx.x] += sq[threadIdx.x + st];
    }
    __syncthreads();
  }
  if (threadIdx.x == 0) { ps[b] = ss[0]; pq[b] = sq[0]; }
}

// Partial sums over bf16 NHWC tensor: block b covers 256 pixels, thread t = channel t.
__global__ __launch_bounds__(256) void bn1_partial_kernel(
    const u16* __restrict__ y, float* __restrict__ ps, float* __restrict__ pq) {
  const int b = blockIdx.x;                 // 0..391
  const int c = threadIdx.x;
  const u16* p = y + (size_t)b * 256 * CH + c;
  float s = 0.f, q = 0.f;
  for (int i = 0; i < 256; ++i) { float v = bf2f(p[(size_t)i * CH]); s += v; q += v * v; }
  ps[(size_t)b * CH + c] = s;
  pq[(size_t)b * CH + c] = q;
}

// Per-channel: scale = gamma*rsqrt(var+eps), bias = beta - mean*scale.
__global__ __launch_bounds__(256) void bn_finalize_kernel(
    const float* __restrict__ ps, const float* __restrict__ pq,
    const float* __restrict__ gamma, const float* __restrict__ beta,
    float* __restrict__ scale, float* __restrict__ bias, int npart) {
  const int c = threadIdx.x;
  float s = 0.f, q = 0.f;
  for (int i = 0; i < npart; ++i) { s += ps[(size_t)i * CH + c]; q += pq[(size_t)i * CH + c]; }
  const float inv = 1.0f / (float)NPIX;
  const float mean = s * inv;
  const float var = q * inv - mean * mean;
  const float sc = gamma[c] * rsqrtf(var + BN_EPS);
  scale[c] = sc;
  bias[c] = beta[c] - mean * sc;
}

// --------------------------- weight quantization ---------------------------

__global__ __launch_bounds__(256) void wabs_partial_kernel(
    const float* __restrict__ w, float* __restrict__ pe) {
  __shared__ float ss[256];
  const int b = blockIdx.x;                 // co
  const float* p = w + (size_t)b * KDIM;
  float s = 0.f;
  for (int i = threadIdx.x; i < KDIM; i += 256) s += fabsf(p[i]);
  ss[threadIdx.x] = s;
  __syncthreads();
  for (int st = 128; st > 0; st >>= 1) {
    if ((int)threadIdx.x < st) ss[threadIdx.x] += ss[threadIdx.x + st];
    __syncthreads();
  }
  if (threadIdx.x == 0) pe[b] = ss[0];
}

__global__ __launch_bounds__(64) void e_finalize_kernel(
    const float* __restrict__ pe0, const float* __restrict__ pe1, float* __restrict__ ev) {
  if (threadIdx.x < 2) {
    const float* pe = threadIdx.x ? pe1 : pe0;
    float s = 0.f;
    for (int i = 0; i < CH; ++i) s += pe[i];
    ev[threadIdx.x] = s / (float)(CH * KDIM);
  }
}

// wq[co][k] bf16 with k = tap*256 + c ; wq = E * sign(w) (sign(0)=0).
__global__ __launch_bounds__(256) void quantize_kernel(
    const float* __restrict__ w, const float* __restrict__ ev, int eidx,
    u16* __restrict__ wq) {
  const float E = ev[eidx];
  const int total = CH * KDIM;
  for (int i = blockIdx.x * 256 + threadIdx.x; i < total; i += gridDim.x * 256) {
    const int co = i / KDIM;
    const int r = i - co * KDIM;
    const int tap = r >> 8;
    const int c = r & 255;
    const float v = w[((size_t)co * CH + c) * 9 + tap];
    const float q = (v > 0.f) ? E : (v < 0.f ? -E : 0.f);
    wq[i] = f2bf(q);
  }
}

// ------------------ BN+ReLU fused with layout transforms -------------------

// NCHW fp32 -> NHWC bf16 with y = relu(x*scale_c + bias_c). LDS-tiled transpose.
__global__ __launch_bounds__(256) void bn0_act_kernel(
    const float* __restrict__ x, const float* __restrict__ scale,
    const float* __restrict__ bias, u16* __restrict__ act) {
  __shared__ float tile[32][33];
  const int n = blockIdx.z;
  const int hw0 = blockIdx.x * 32;
  const int c0 = blockIdx.y * 32;
  const int tx = threadIdx.x, ty = threadIdx.y;   // (32,8)
#pragma unroll
  for (int i = 0; i < 4; ++i) {
    const int c = c0 + ty + i * 8;
    float v = x[((size_t)n * CH + c) * HW + hw0 + tx];
    v = v * scale[c] + bias[c];
    tile[ty + i * 8][tx] = fmaxf(v, 0.f);         // tile[c_local][hw_local]
  }
  __syncthreads();
#pragma unroll
  for (int i = 0; i < 4; ++i) {
    const int hw = hw0 + ty + i * 8;
    act[((size_t)n * HW + hw) * CH + c0 + tx] = f2bf(tile[tx][ty + i * 8]);
  }
}

// Elementwise NHWC bf16 -> bf16: a = relu(y*scale_c + bias_c).
__global__ __launch_bounds__(256) void bn1_act_kernel(
    const u16* __restrict__ y, const float* __restrict__ scale,
    const float* __restrict__ bias, u16* __restrict__ a) {
  const size_t i = (size_t)blockIdx.x * 256 + threadIdx.x;
  if (i < (size_t)NPIX * CH) {
    const int c = (int)(i & 255);
    const float v = bf2f(y[i]) * scale[c] + bias[c];
    a[i] = f2bf(fmaxf(v, 0.f));
  }
}

// NHWC fp32 -> NCHW fp32 with residual add from x (NCHW).
__global__ __launch_bounds__(256) void epilogue_kernel(
    const float* __restrict__ y2, const float* __restrict__ x, float* __restrict__ out) {
  __shared__ float tile[32][33];
  const int n = blockIdx.z;
  const int hw0 = blockIdx.x * 32;
  const int c0 = blockIdx.y * 32;
  const int tx = threadIdx.x, ty = threadIdx.y;
#pragma unroll
  for (int i = 0; i < 4; ++i) {
    const int hw = hw0 + ty + i * 8;
    tile[ty + i * 8][tx] = y2[((size_t)n * HW + hw) * CH + c0 + tx]; // tile[hw_l][c_l]
  }
  __syncthreads();
#pragma unroll
  for (int i = 0; i < 4; ++i) {
    const int c = c0 + ty + i * 8;
    const size_t idx = ((size_t)n * CH + c) * HW + hw0 + tx;
    out[idx] = tile[tx][ty + i * 8] + x[idx];
  }
}

// --------------------------- WMMA implicit-GEMM conv -----------------------
// D[co, pix] = sum_k Wq[co, k] * Act[k, pix],  k = tap*256 + c.
// Block: 256 threads = 8 waves; tile 128(co) x 128(pix); K step 32.
// Wave grid 4x2: wv_m covers 32 co rows (2 A-frags), wv_n covers 64 pixels
// (4 B-frags) -> 8 accumulators, 8 WMMAs per K-step per wave.
// LDS double buffered: async-stage slice t+1 while computing slice t.

__device__ __forceinline__ v16bf load_frag(const u16* __restrict__ row, int hb) {
  // Per-lane fragment per CDNA5 16-bit A/B layout:
  //   lanes 0-15  hold K = {0..7, 16..23}   (hb = 0)
  //   lanes 16-31 hold K = {8..15, 24..31}  (hb = 8)
  union { uint4 u[2]; v16bf v; } t;
  t.u[0] = *(const uint4*)(row + hb);        // halfs [hb, hb+8)
  t.u[1] = *(const uint4*)(row + 16 + hb);   // halfs [16+hb, 24+hb)
  return t.v;
}

// Stage one K-slice: weights rows [co0,co0+128) x k[kk,kk+32) and activations
// rows [pix0,pix0+128) x 32 channels of tap(kk). Each thread moves 32B per
// matrix (2x b128). srow = tid>>1 (0..127), shalf = tid&1 (16-half segment).
__device__ __forceinline__ void stage_slice(
    const u16* __restrict__ act, const u16* __restrict__ wq,
    u16* __restrict__ wt, u16* __restrict__ at,
    int co0, int kk, int srow, int shalf,
    int n, int h, int w) {
  const int tap = kk >> 8;
  const int c0 = kk & 255;
  // weights
  {
    const u16* src = wq + (size_t)(co0 + srow) * KDIM + kk + shalf * 16;
    __builtin_prefetch(src + BK, 0, 0);      // global_prefetch next K slice
    u16* dst = &wt[srow * LDS_K + shalf * 16];
    cp_async16(src, dst);
    cp_async16(src + 8, dst + 8);
  }
  // activations (zero pad outside the image)
  {
    const int dh = tap / 3 - 1;
    const int dw = tap - (tap / 3) * 3 - 1;
    const int ih = h + dh;
    const int iw = w + dw;
    u16* dst = &at[srow * LDS_K + shalf * 16];
    if ((unsigned)ih < IMGH && (unsigned)iw < IMGW) {
      const u16* src = act + ((size_t)(n * HW + ih * IMGW + iw) * CH + c0 + shalf * 16);
      cp_async16(src, dst);
      cp_async16(src + 8, dst + 8);
    } else {
      const uint4 z = make_uint4(0u, 0u, 0u, 0u);
      ((uint4*)dst)[0] = z;
      ((uint4*)dst)[1] = z;
    }
  }
}

__global__ __launch_bounds__(256) void conv_wmma_kernel(
    const u16* __restrict__ act,    // NHWC bf16 [NPIX][CH]
    const u16* __restrict__ wq,     // bf16 [CH][KDIM]
    u16* __restrict__ out_bf16,     // NHWC bf16 (conv1) or nullptr
    float* __restrict__ out_f32) {  // NHWC fp32 (conv2) or nullptr
  __shared__ u16 Wt[2][BM * LDS_K];        // 2 x 10240 B
  __shared__ u16 At[2][BN_TILE * LDS_K];   // 2 x 10240 B

  const int tid = threadIdx.x;
  const int lane = tid & 31;
  const int wv = tid >> 5;
  const int wv_m = wv >> 1;                  // 0..3 -> 32 co rows each
  const int wv_n = wv & 1;                   // 0..1 -> 64 pixels each
  const int hb = (lane & 16) ? 8 : 0;
  const int lrow = lane & 15;

  const int pix0 = blockIdx.x * BN_TILE;     // NPIX / 128 = 784 tiles
  const int co0 = blockIdx.y * BM;           // CH / 128 = 2 tiles

  // staging coords: 128 rows x 2 segments of 16 halfs (32B)
  const int srow = tid >> 1;
  const int shalf = tid & 1;

  const int pg = pix0 + srow;
  const int n = pg / HW;
  const int hwp = pg - n * HW;
  const int h = hwp / IMGW;
  const int w = hwp - h * IMGW;

  v8f acc[2][4];
#pragma unroll
  for (int i = 0; i < 2; ++i)
#pragma unroll
    for (int j = 0; j < 4; ++j) acc[i][j] = (v8f){};

  // prologue: stage slice 0 into buffer 0
  stage_slice(act, wq, Wt[0], At[0], co0, 0, srow, shalf, n, h, w);
  async_fence();
  __syncthreads();

  for (int it = 0; it < KDIM / BK; ++it) {
    const int cur = it & 1;
    const int kk_next = (it + 1) * BK;
    if (kk_next < KDIM) {
      stage_slice(act, wq, Wt[1 - cur], At[1 - cur], co0, kk_next, srow, shalf, n, h, w);
    }

    v16bf afrag[2], bfrag[4];
#pragma unroll
    for (int i = 0; i < 2; ++i)
      afrag[i] = load_frag(&Wt[cur][(wv_m * 32 + i * 16 + lrow) * LDS_K], hb);
#pragma unroll
    for (int j = 0; j < 4; ++j)
      bfrag[j] = load_frag(&At[cur][(wv_n * 64 + j * 16 + lrow) * LDS_K], hb);

#pragma unroll
    for (int j = 0; j < 4; ++j) {
#pragma unroll
      for (int i = 0; i < 2; ++i) {
        acc[i][j] = __builtin_amdgcn_wmma_f32_16x16x32_bf16(
            false, afrag[i], false, bfrag[j], (short)0, acc[i][j], false, false);
      }
    }
    async_fence();   // staged slice t+1 complete before the barrier
    __syncthreads();
  }

  // C layout: lane holds col = lane&15, rows m = hb + r (r=0..7) -> 8 consecutive
  // output channels per lane -> fully vectorized stores.
#pragma unroll
  for (int i = 0; i < 2; ++i) {
    const int co_base = co0 + wv_m * 32 + i * 16 + hb;
#pragma unroll
    for (int j = 0; j < 4; ++j) {
      const v8f a = acc[i][j];
      const int pix = pix0 + wv_n * 64 + j * 16 + lrow;
      if (out_bf16 != nullptr) {
        union { uint4 u; u16 hh[8]; } pk;
#pragma unroll
        for (int r = 0; r < 8; ++r) pk.hh[r] = f2bf(a[r]);
        *(uint4*)&out_bf16[(size_t)pix * CH + co_base] = pk.u;
      } else {
        float* dst = out_f32 + (size_t)pix * CH + co_base;
        *(float4*)dst = make_float4(a[0], a[1], a[2], a[3]);
        *(float4*)(dst + 4) = make_float4(a[4], a[5], a[6], a[7]);
      }
    }
  }
}

// --------------------------------- launch ----------------------------------

extern "C" void kernel_launch(void* const* d_in, const int* in_sizes, int n_in,
                              void* d_out, int out_size, void* d_ws, size_t ws_size,
                              hipStream_t stream) {
  (void)in_sizes; (void)n_in; (void)out_size; (void)ws_size;

  const float* x      = (const float*)d_in[0];
  const float* gamma0 = (const float*)d_in[1];
  const float* beta0  = (const float*)d_in[2];
  const float* w0     = (const float*)d_in[3];
  const float* gamma1 = (const float*)d_in[4];
  const float* beta1  = (const float*)d_in[5];
  const float* w1     = (const float*)d_in[6];
  float* out = (float*)d_out;

  // workspace carve-up (256B aligned)
  char* ws = (char*)d_ws;
  size_t off = 0;
  auto carve = [&](size_t bytes) -> char* {
    char* p = ws + off;
    off = (off + bytes + 255) & ~(size_t)255;
    return p;
  };
  u16*   ACT = (u16*)  carve((size_t)NPIX * CH * 2);   // 51.4 MB bf16 NHWC (a0, then a1)
  u16*   Y1  = (u16*)  carve((size_t)NPIX * CH * 2);   // 51.4 MB bf16 NHWC
  float* Y2  = (float*)carve((size_t)NPIX * CH * 4);   // 102.8 MB fp32 NHWC
  u16*   WQ0 = (u16*)  carve((size_t)CH * KDIM * 2);   // 1.18 MB
  u16*   WQ1 = (u16*)  carve((size_t)CH * KDIM * 2);
  float* PS0 = (float*)carve((size_t)BATCH * CH * 4);
  float* PQ0 = (float*)carve((size_t)BATCH * CH * 4);
  float* PS1 = (float*)carve((size_t)392 * CH * 4);
  float* PQ1 = (float*)carve((size_t)392 * CH * 4);
  float* PE0 = (float*)carve((size_t)CH * 4);
  float* PE1 = (float*)carve((size_t)CH * 4);
  float* EV  = (float*)carve(2 * 4);
  float* SC0 = (float*)carve((size_t)CH * 4);
  float* BI0 = (float*)carve((size_t)CH * 4);
  float* SC1 = (float*)carve((size_t)CH * 4);
  float* BI1 = (float*)carve((size_t)CH * 4);

  // 1) BN0 stats
  bn0_partial_kernel<<<BATCH * CH, 256, 0, stream>>>(x, PS0, PQ0);
  bn_finalize_kernel<<<1, 256, 0, stream>>>(PS0, PQ0, gamma0, beta0, SC0, BI0, BATCH);

  // 2) weight quantization (E = mean|w|, wq = E*sign(w), bf16, tap-major K)
  wabs_partial_kernel<<<CH, 256, 0, stream>>>(w0, PE0);
  wabs_partial_kernel<<<CH, 256, 0, stream>>>(w1, PE1);
  e_finalize_kernel<<<1, 64, 0, stream>>>(PE0, PE1, EV);
  quantize_kernel<<<1024, 256, 0, stream>>>(w0, EV, 0, WQ0);
  quantize_kernel<<<1024, 256, 0, stream>>>(w1, EV, 1, WQ1);

  // 3) a0 = relu(BN0(x)) as bf16 NHWC
  bn0_act_kernel<<<dim3(HW / 32, CH / 32, BATCH), dim3(32, 8), 0, stream>>>(x, SC0, BI0, ACT);

  // 4) conv1 (WMMA) -> Y1 bf16 NHWC
  conv_wmma_kernel<<<dim3(NPIX / BN_TILE, CH / BM), 256, 0, stream>>>(ACT, WQ0, Y1, nullptr);

  // 5) BN1 stats + a1 = relu(BN1(y1)) (reuse ACT buffer)
  bn1_partial_kernel<<<392, 256, 0, stream>>>(Y1, PS1, PQ1);
  bn_finalize_kernel<<<1, 256, 0, stream>>>(PS1, PQ1, gamma1, beta1, SC1, BI1, 392);
  bn1_act_kernel<<<(unsigned)((size_t)NPIX * CH / 256), 256, 0, stream>>>(Y1, SC1, BI1, ACT);

  // 6) conv2 (WMMA) -> Y2 fp32 NHWC
  conv_wmma_kernel<<<dim3(NPIX / BN_TILE, CH / BM), 256, 0, stream>>>(ACT, WQ1, nullptr, Y2);

  // 7) NHWC -> NCHW + residual
  epilogue_kernel<<<dim3(HW / 32, CH / 32, BATCH), dim3(32, 8), 0, stream>>>(Y2, x, out);
}
